// Correlation_5428838662350
// MI455X (gfx1250) — matrix operand order repaired and checked by reading, not compiled, not run on hardware
//
#include <hip/hip_runtime.h>

typedef __attribute__((ext_vector_type(16))) _Float16 v16h;
typedef __attribute__((ext_vector_type(8)))  float    v8f;

#define NW    8            // waves per block (wave32 -> 256 threads)
#define CH    16           // channels per chunk
#define NCC   (256 / CH)   // channel chunks
#define ROWU  64           // uints per row copy (128 f16 slots; 112 needed)

__global__ void Corr_zero_kernel(float* out, int n) {
    int i = blockIdx.x * blockDim.x + threadIdx.x;
    if (i < n) out[i] = 0.0f;
}

union Frag16 { unsigned int u[8]; v16h h; };

// Two f32 -> packed f16x2 (v_cvt_f16_f32 .l/.h halves, RTE)
__device__ __forceinline__ unsigned int pack_f16(float f0, float f1) {
    _Float16 h0 = (_Float16)f0, h1 = (_Float16)f1;
    unsigned short u0, u1;
    __builtin_memcpy(&u0, &h0, 2);
    __builtin_memcpy(&u1, &h1, 2);
    return (unsigned int)u0 | ((unsigned int)u1 << 16);
}

// Store one padded search row (f16) into the dual-copy buffer.
// copy0 uint d  = padded elements {2d, 2d+1}        (even-base copy)
// copy1 uint d  = padded elements {2d+1, 2d+2}      (odd-base copy, via lane shuffle)
// Lane L owns padded elements {16+2L, 17+2L} (real cols 0..63 at padded 16..79).
__device__ __forceinline__ void store_row(unsigned int* row0u, unsigned int* row1u,
                                          unsigned int pk, int lane) {
    row0u[8 + lane] = pk;
    unsigned int pk_next = (unsigned int)__shfl_down((int)pk, 1, 32);
    pk_next = (lane == 31) ? 0u : pk_next;            // padded[80] = 0
    row1u[8 + lane] = (pk >> 16) | (pk_next << 16);   // {padded 17+2L, 18+2L}
    if (lane == 0) row1u[7] = pk << 16;               // {padded 15 (=0), 16}
}

__global__ void __launch_bounds__(NW * 32)
Corr_wmma_kernel(const float* __restrict__ query,
                 const float* __restrict__ search,
                 float* __restrict__ out) {
    // Query chunk pre-swizzled: lane's B fragment = 8 contiguous uints (16 f16).
    __shared__ unsigned int qswz[CH * 2 * 32 * 8];        // 32 KB
    __shared__ float        accs[65 * 16];                // 65 y-rows x 16 x
    __shared__ unsigned int rowbuf[NW][2][2][ROWU];       // [wave][slot][copy], 8 KB

    const int tid  = threadIdx.x;
    const int lane = tid & 31;
    const int wv   = tid >> 5;

    int bi = blockIdx.x;
    const int cc = bi % NCC;  bi /= NCC;
    const int xt = bi % 5;    bi /= 5;
    const int b  = bi;
    const int x0 = xt * 16;
    const int c0 = cc * CH;

    // ---- init LDS ----
    for (int i = tid; i < 65 * 16; i += NW * 32) accs[i] = 0.0f;
    {
        unsigned int* rb = &rowbuf[0][0][0][0];
        for (int i = tid; i < NW * 2 * 2 * ROWU; i += NW * 32) rb[i] = 0u;
    }
    // ---- stage + swizzle query chunk (f32 -> packed f16) ----
    // B layout (ISA 7.12.2): column n = lane%16; lanes0-15 K{0-7,16-23}, lanes16-31 K{8-15,24-31}.
    for (int idx = tid; idx < CH * 2 * 32 * 8; idx += NW * 32) {
        int p = idx & 7;             // uint slot -> K pair {2p, 2p+1}
        int L = (idx >> 3) & 31;     // lane
        int t = (idx >> 8) & 1;      // n-tile (i half)
        int c = idx >> 9;            // channel within chunk
        int irow = (L & 15) + 16 * t;          // N dim: query row i
        int k0   = 2 * p + 8 * (L >> 4) + 8 * (p >> 2);      // K dim: query col j
        const float* qp = query + (((size_t)(b * 256 + c0 + c)) * 32 + irow) * 32 + k0;
        qswz[idx] = pack_f16(qp[0], qp[1]);
    }
    __syncthreads();

    const int m  = lane & 15;            // A row (x within tile)
    const int kb = (lane >> 4) << 3;     // A K base: lanes0-15 K{0-7,16-23}, lanes16-31 K{8-15,24-31}
    const float* sbase = search + ((size_t)(b * 256 + c0)) * 64 * 64;

    for (int r = wv; r < 64; r += NW) {  // uniform trip count per wave
        // stage channel 0 into slot 0
        {
            const float* src = sbase + (size_t)r * 64;
            unsigned int pk = pack_f16(src[2 * lane], src[2 * lane + 1]);
            store_row(rowbuf[wv][0][0], rowbuf[wv][0][1], pk, lane);
        }
        v8f acc0 = {};   // D[x, i=0..15]
        v8f acc1 = {};   // D[x, i=16..31]
        for (int c = 0; c < CH; ++c) {
            const int cur = c & 1, nxt = cur ^ 1;

            // issue next channel's global row load early (overlaps WMMA below)
            float nf0 = 0.0f, nf1 = 0.0f;
            if (c + 1 < CH) {
                const float* nsrc = sbase + (size_t)(c + 1) * 4096 + (size_t)r * 64;
                nf0 = nsrc[2 * lane];
                nf1 = nsrc[2 * lane + 1];
            }
            // prefetch this channel's row for the next r iteration
            if (r + NW < 64)
                __builtin_prefetch(sbase + (size_t)c * 4096 + (size_t)(r + NW) * 64
                                   + 2 * lane, 0, 1);

            // ---- A fragment: sliding window Prow[x0+m+K] from slot 'cur' ----
            const unsigned int* r0u = rowbuf[wv][cur][0];
            const unsigned int* r1u = rowbuf[wv][cur][1];
            Frag16 a, b0f, b1f;
            int s1 = x0 + m + kb;
            {
                const unsigned int* p = (s1 & 1) ? (r1u + ((s1 - 1) >> 1))
                                                 : (r0u + (s1 >> 1));
                a.u[0] = p[0]; a.u[1] = p[1]; a.u[2] = p[2]; a.u[3] = p[3];
            }
            int s2 = s1 + 16;
            {
                const unsigned int* p = (s2 & 1) ? (r1u + ((s2 - 1) >> 1))
                                                 : (r0u + (s2 >> 1));
                a.u[4] = p[0]; a.u[5] = p[1]; a.u[6] = p[2]; a.u[7] = p[3];
            }
            // ---- B fragments: one 32B aligned LDS read each ----
            __builtin_memcpy(b0f.u, (const unsigned int*)__builtin_assume_aligned(
                                 &qswz[((c * 2 + 0) * 32 + lane) * 8], 16), 32);
            __builtin_memcpy(b1f.u, (const unsigned int*)__builtin_assume_aligned(
                                 &qswz[((c * 2 + 1) * 32 + lane) * 8], 16), 32);

            acc0 = __builtin_amdgcn_wmma_f32_16x16x32_f16(
                false, a.h, false, b0f.h, (short)0, acc0, false, false);
            acc1 = __builtin_amdgcn_wmma_f32_16x16x32_f16(
                false, a.h, false, b1f.h, (short)0, acc1, false, false);

            // ---- stage next channel's row into the other slot (after WMMAs) ----
            if (c + 1 < CH) {
                unsigned int pk = pack_f16(nf0, nf1);
                store_row(rowbuf[wv][nxt][0], rowbuf[wv][nxt][1], pk, lane);
            }
        }
        // ---- scatter D[x,i] -> accs[y=(r+16)-i][x]  (C/D layout: M=g / g+8) ----
        const int xadd = (lane >> 4) << 3;
        const int i0   = lane & 15;
        const int y0   = r + 16 - i0;
        for (int g = 0; g < 8; ++g) {
            int xl = g + xadd;
            if ((unsigned)y0 < 65u)       atomicAdd(&accs[y0 * 16 + xl], acc0[g]);
            int y1 = y0 - 16;             // i = i0 + 16
            if ((unsigned)y1 < 65u)       atomicAdd(&accs[y1 * 16 + xl], acc1[g]);
        }
    }
    __syncthreads();

    // ---- merge this channel-chunk's partial into global output ----
    for (int idx = tid; idx < 65 * 16; idx += NW * 32) {
        int y  = idx >> 4;
        int xg = x0 + (idx & 15);
        if (xg < 65)
            atomicAdd(&out[(size_t)b * 65 * 65 + y * 65 + xg], accs[idx]);
    }
}

extern "C" void kernel_launch(void* const* d_in, const int* in_sizes, int n_in,
                              void* d_out, int out_size, void* d_ws, size_t ws_size,
                              hipStream_t stream) {
    const float* query  = (const float*)d_in[0];   // [16,256,32,32] f32
    const float* search = (const float*)d_in[1];   // [16,256,64,64] f32
    float* out = (float*)d_out;                    // [16,1,65,65]   f32

    Corr_zero_kernel<<<(out_size + 255) / 256, 256, 0, stream>>>(out, out_size);

    dim3 grid(16 * 5 * NCC);                       // batch x x-tiles x channel-chunks
    Corr_wmma_kernel<<<grid, NW * 32, 0, stream>>>(query, search, out);
}
